// PhysicsBiasedBlock_6098853560435
// MI455X (gfx1250) — matrix operand order repaired and checked by reading, not compiled, not run on hardware
//
#include <hip/hip_runtime.h>
#include <hip/hip_bf16.h>
#include <math.h>

// ---------------------------------------------------------------------------
// Types for CDNA5 WMMA (wave32): v_wmma_f32_16x16x32_bf16
// ---------------------------------------------------------------------------
typedef __bf16 bf16_t;
typedef __attribute__((ext_vector_type(16))) __bf16 v16bf;
typedef __attribute__((ext_vector_type(8)))  __bf16 bf16x8;
typedef __attribute__((ext_vector_type(8)))  float  v8f;
typedef __attribute__((ext_vector_type(4)))  unsigned u32x4;
typedef __attribute__((ext_vector_type(8)))  unsigned u32x8;

union ABFrag { v16bf v; bf16x8 h[2]; };

__device__ __forceinline__ bf16_t to_bf16(float f) {
  unsigned u = __float_as_uint(f);
  u += 0x7fffu + ((u >> 16) & 1u);   // round-to-nearest-even
  unsigned short s = (unsigned short)(u >> 16);
  bf16_t r;
  __builtin_memcpy(&r, &s, 2);
  return r;
}

// Problem constants
#define BB   4
#define NN   1024
#define CC   768
#define HH   12
#define DD   64
#define HID  3072
#define MTOT (BB * NN)      // 4096 rows
#define SCALE 0.125f        // D^-0.5
#define EPS   1e-6f

// ---------------------------------------------------------------------------
// Tensor Data Mover: issue a 2D tile load (bf16 elements) into LDS.
// D# per CDNA5 ISA 8.3/8.4.
// ---------------------------------------------------------------------------
__device__ __forceinline__
void tdm_load_2d(const void* gptr, unsigned lds_off,
                 unsigned tile_w, unsigned tile_h, unsigned stride_elems) {
  unsigned long long ga = (unsigned long long)(uintptr_t)gptr;
  u32x4 g0;
  g0[0] = 1u;                                            // count=1 (valid user D#)
  g0[1] = lds_off;                                       // LDS byte address
  g0[2] = (unsigned)ga;                                  // global_addr[31:0]
  g0[3] = (unsigned)((ga >> 32) & 0x01ffffffu) | (2u << 30); // addr[56:32] | type=2
  u32x8 g1;
  g1[0] = 1u << 16;                                      // data_size = 2 bytes
  g1[1] = (tile_w & 0xffffu) << 16;                      // tensor_dim0[15:0]
  g1[2] = ((tile_h & 0xffffu) << 16) | (tile_w >> 16);   // tensor_dim1 lo | dim0 hi
  g1[3] = (tile_w << 16) | (tile_h >> 16);               // tile_dim0 | tensor_dim1 hi
  g1[4] = tile_h & 0xffffu;                              // tile_dim1 (tile_dim2 = 0)
  g1[5] = stride_elems;                                  // tensor_dim0_stride[31:0]
  g1[6] = 0u;
  g1[7] = 0u;
  asm volatile("tensor_load_to_lds %0, %1" :: "s"(g0), "s"(g1) : "memory");
}

// ---------------------------------------------------------------------------
// Weight convert + transpose:  W[K,N] f32  ->  Wt[N,K] bf16  (K contiguous)
// ---------------------------------------------------------------------------
__global__ __launch_bounds__(256)
void cvt_transpose_kernel(const float* __restrict__ W, bf16_t* __restrict__ Wt,
                          int K, int Nn) {
  int idx = blockIdx.x * 256 + threadIdx.x;
  if (idx >= K * Nn) return;
  int k = idx / Nn, n = idx - k * Nn;
  Wt[(size_t)n * K + k] = to_bf16(W[idx]);
}

// ---------------------------------------------------------------------------
// LayerNorm over C=768, output bf16.  One block (256 threads) per row.
// ---------------------------------------------------------------------------
__global__ __launch_bounds__(256)
void ln_bf16_kernel(const float* __restrict__ x, const float* __restrict__ w,
                    const float* __restrict__ b, bf16_t* __restrict__ out) {
  __shared__ float red[16];
  const int row = blockIdx.x;
  const float* xr = x + (size_t)row * CC;
  const int t = threadIdx.x;
  float v0 = xr[t], v1 = xr[t + 256], v2 = xr[t + 512];
  float s1 = v0 + v1 + v2;
  float s2 = v0 * v0 + v1 * v1 + v2 * v2;
#pragma unroll
  for (int m = 16; m; m >>= 1) {
    s1 += __shfl_xor(s1, m, 32);
    s2 += __shfl_xor(s2, m, 32);
  }
  if ((t & 31) == 0) { red[t >> 5] = s1; red[8 + (t >> 5)] = s2; }
  __syncthreads();
  float sum = 0.f, sumsq = 0.f;
#pragma unroll
  for (int i = 0; i < 8; ++i) { sum += red[i]; sumsq += red[8 + i]; }
  const float mu = sum * (1.0f / CC);
  const float var = sumsq * (1.0f / CC) - mu * mu;
  const float rs = rsqrtf(var + EPS);
  bf16_t* orow = out + (size_t)row * CC;
#pragma unroll
  for (int i = 0; i < 3; ++i) {
    int c = t + i * 256;
    float v = (i == 0) ? v0 : (i == 1) ? v1 : v2;
    orow[c] = to_bf16((v - mu) * rs * w[c] + b[c]);
  }
}

// ---------------------------------------------------------------------------
// TDM-staged, double-buffered GEMM mainloop (macro so the __shared__ arrays
// are referenced directly with constant buffer indices -> ds_load_b128).
// Block = 256 threads (8 waves); block tile = 128 rows x 64 cols.
// Each TDM stage stages K=64 (A: 128x64 = 16KB, B: 64x64 = 8KB), ping-pong.
// GEMM_COMPUTE: issue ALL 20 ds_load_b128 first, then 8 back-to-back WMMAs.
// ---------------------------------------------------------------------------
#define GEMM_LDS_DECL                                                         \
  __shared__ __align__(16) bf16_t lA[2][128 * 64];                            \
  __shared__ __align__(16) bf16_t lB[2][64 * 64];

#define GEMM_COMPUTE(buf)                                                     \
  {                                                                           \
    ABFrag afr[2];                                                            \
    ABFrag bfr[2][4];                                                         \
    _Pragma("unroll")                                                         \
    for (int ks = 0; ks < 2; ++ks) {                                          \
      const int ao = (wave * 16 + lm) * 64 + ks * 32;                         \
      afr[ks].h[0] = *(const bf16x8*)(&lA[buf][ao + half * 16]);              \
      afr[ks].h[1] = *(const bf16x8*)(&lA[buf][ao + half * 16 + 8]);          \
      _Pragma("unroll")                                                       \
      for (int tt = 0; tt < 4; ++tt) {                                        \
        const int bo = (tt * 16 + lm) * 64 + ks * 32 + half * 16;             \
        bfr[ks][tt].h[0] = *(const bf16x8*)(&lB[buf][bo]);                    \
        bfr[ks][tt].h[1] = *(const bf16x8*)(&lB[buf][bo + 8]);                \
      }                                                                       \
    }                                                                         \
    _Pragma("unroll")                                                         \
    for (int ks = 0; ks < 2; ++ks)                                            \
      _Pragma("unroll")                                                       \
      for (int tt = 0; tt < 4; ++tt)                                          \
        acc[tt] = __builtin_amdgcn_wmma_f32_16x16x32_bf16(                    \
            false, afr[ks].v, false, bfr[ks][tt].v, (short)0, acc[tt],        \
            false, false);                                                    \
  }

#define GEMM_MAINLOOP(Aptr, lda_, Wt_, ldk_, K_)                              \
  const int wave = threadIdx.x >> 5;                                          \
  const int lane = threadIdx.x & 31;                                          \
  const int half = lane >> 4, lm = lane & 15;                                 \
  const int Ssteps = (K_) >> 6;                                               \
  const bf16_t* agbase = (Aptr) + (size_t)(blockIdx.y * 128) * (lda_);        \
  const bf16_t* bgbase = (Wt_) + (size_t)(blockIdx.x * 64) * (ldk_);          \
  if (wave == 0) {                                                            \
    tdm_load_2d(agbase, (unsigned)(uintptr_t)&lA[0][0], 64u, 128u,            \
                (unsigned)(lda_));                                            \
    tdm_load_2d(bgbase, (unsigned)(uintptr_t)&lB[0][0], 64u, 64u,             \
                (unsigned)(ldk_));                                            \
  }                                                                           \
  for (int s = 0; s < Ssteps; s += 2) {                                       \
    if (wave == 0) {                                                          \
      if (s + 1 < Ssteps) {                                                   \
        tdm_load_2d(agbase + (size_t)(s + 1) * 64,                            \
                    (unsigned)(uintptr_t)&lA[1][0], 64u, 128u,                \
                    (unsigned)(lda_));                                        \
        tdm_load_2d(bgbase + (size_t)(s + 1) * 64,                            \
                    (unsigned)(uintptr_t)&lB[1][0], 64u, 64u,                 \
                    (unsigned)(ldk_));                                        \
        __builtin_amdgcn_s_wait_tensorcnt(2);                                 \
      } else {                                                                \
        __builtin_amdgcn_s_wait_tensorcnt(0);                                 \
      }                                                                       \
    }                                                                         \
    __syncthreads();                                                          \
    GEMM_COMPUTE(0)                                                           \
    __syncthreads();                                                          \
    if (s + 1 < Ssteps) {                                                     \
      if (wave == 0) {                                                        \
        if (s + 2 < Ssteps) {                                                 \
          tdm_load_2d(agbase + (size_t)(s + 2) * 64,                          \
                      (unsigned)(uintptr_t)&lA[0][0], 64u, 128u,              \
                      (unsigned)(lda_));                                      \
          tdm_load_2d(bgbase + (size_t)(s + 2) * 64,                          \
                      (unsigned)(uintptr_t)&lB[0][0], 64u, 64u,               \
                      (unsigned)(ldk_));                                      \
          __builtin_amdgcn_s_wait_tensorcnt(2);                               \
        } else {                                                              \
          __builtin_amdgcn_s_wait_tensorcnt(0);                               \
        }                                                                     \
      }                                                                       \
      __syncthreads();                                                        \
      GEMM_COMPUTE(1)                                                         \
      __syncthreads();                                                        \
    }                                                                         \
  }

// ---------------------------------------------------------------------------
// QKV GEMM: a1[4096,768] @ qkv_w -> scatter q(pre-scaled)/k bf16 [B,H,N,D],
// v transposed bf16 [B,H,D,N].
// ---------------------------------------------------------------------------
__global__ __launch_bounds__(256)
void gemm_qkv_kernel(const bf16_t* __restrict__ A, const bf16_t* __restrict__ Wt,
                     const float* __restrict__ bias,
                     bf16_t* __restrict__ qb, bf16_t* __restrict__ kb,
                     bf16_t* __restrict__ vT) {
  GEMM_LDS_DECL
  v8f acc[4] = {};
  GEMM_MAINLOOP(A, CC, Wt, CC, CC)
  const int row0 = blockIdx.y * 128 + wave * 16;
  const int col0 = blockIdx.x * 64;
#pragma unroll
  for (int tt = 0; tt < 4; ++tt) {
    const int col = col0 + tt * 16 + lm;
    const int i = col / CC, rem = col - i * CC;
    const int h = rem >> 6, d = rem & 63;
    const float bbv = bias[col];
#pragma unroll
    for (int r = 0; r < 8; ++r) {
      const int row = row0 + r + half * 8;
      const int b = row >> 10, n = row & 1023;
      const float val = acc[tt][r] + bbv;
      const size_t hn = (size_t)(b * HH + h) * NN + n;
      if (i == 0)      qb[hn * DD + d] = to_bf16(val * SCALE);
      else if (i == 1) kb[hn * DD + d] = to_bf16(val);
      else             vT[((size_t)(b * HH + h) * DD + d) * NN + n] = to_bf16(val);
    }
  }
}

// ---------------------------------------------------------------------------
// Flash-style biased attention, software-pipelined:
//  - K fragments for block kb0+32 preloaded during block kb0
//  - V fragments issued before softmax VALU work so latency hides
// ---------------------------------------------------------------------------
__global__ __launch_bounds__(128)
void attn_kernel(const bf16_t* __restrict__ qb, const bf16_t* __restrict__ kb,
                 const bf16_t* __restrict__ vT, const float* __restrict__ bias,
                 bf16_t* __restrict__ attn_a) {
  __shared__ __align__(16) bf16_t plds[4][16 * 32];
  const int wave = threadIdx.x >> 5;
  const int lane = threadIdx.x & 31, half = lane >> 4, lm = lane & 15;
  const int bh = blockIdx.y;                 // 0..47
  const int b = bh / HH, h = bh - b * HH;
  const int q0 = blockIdx.x * 64 + wave * 16;

  const bf16_t* qbase = qb + (size_t)bh * NN * DD;
  const bf16_t* kbase = kb + (size_t)bh * NN * DD;
  const bf16_t* vbase = vT + (size_t)bh * DD * NN;
  const float*  bbase = bias + (size_t)b * NN * NN;

  ABFrag qa[2];
  {
    const bf16_t* qrow = qbase + (size_t)(q0 + lm) * DD;
#pragma unroll
    for (int s = 0; s < 2; ++s) {
      qa[s].h[0] = *(const bf16x8*)(qrow + s * 32 + half * 16);
      qa[s].h[1] = *(const bf16x8*)(qrow + s * 32 + half * 16 + 8);
    }
  }

  v8f o[4] = {};
  float mrow[8], lrow[8];
#pragma unroll
  for (int r = 0; r < 8; ++r) { mrow[r] = -3.0e38f; lrow[r] = 0.f; }

  // Preload K fragments for the first key block
  ABFrag kf[2][2];
#pragma unroll
  for (int j = 0; j < 2; ++j) {
    const bf16_t* krow = kbase + (size_t)(j * 16 + lm) * DD;
#pragma unroll
    for (int s = 0; s < 2; ++s) {
      kf[j][s].h[0] = *(const bf16x8*)(krow + s * 32 + half * 16);
      kf[j][s].h[1] = *(const bf16x8*)(krow + s * 32 + half * 16 + 8);
    }
  }

  for (int kb0 = 0; kb0 < NN; kb0 += 32) {
    // ---- S = Q K^T using preloaded K fragments ----
    v8f S[2] = {};
#pragma unroll
    for (int j = 0; j < 2; ++j)
#pragma unroll
      for (int s = 0; s < 2; ++s)
        S[j] = __builtin_amdgcn_wmma_f32_16x16x32_bf16(
            false, qa[s].v, false, kf[j][s].v, (short)0, S[j], false, false);

    // ---- issue V fragment loads for this block (consumed after softmax) ----
    ABFrag vf[4];
#pragma unroll
    for (int tt = 0; tt < 4; ++tt) {
      const bf16_t* vrow = vbase + (size_t)(tt * 16 + lm) * NN + kb0 + half * 16;
      vf[tt].h[0] = *(const bf16x8*)(vrow);
      vf[tt].h[1] = *(const bf16x8*)(vrow + 8);
    }
    // ---- prefetch K fragments for the next block ----
    ABFrag kf2[2][2];
    const bool more = (kb0 + 32) < NN;
    if (more) {
#pragma unroll
      for (int j = 0; j < 2; ++j) {
        const bf16_t* krow = kbase + (size_t)(kb0 + 32 + j * 16 + lm) * DD;
#pragma unroll
        for (int s = 0; s < 2; ++s) {
          kf2[j][s].h[0] = *(const bf16x8*)(krow + s * 32 + half * 16);
          kf2[j][s].h[1] = *(const bf16x8*)(krow + s * 32 + half * 16 + 8);
        }
      }
    }

    // ---- bias + online softmax ----
    float p0[8], p1[8];
#pragma unroll
    for (int r = 0; r < 8; ++r) {
      const int qrow = q0 + r + half * 8;
      const float s0 = S[0][r] + bbase[(size_t)qrow * NN + kb0 + lm];
      const float s1 = S[1][r] + bbase[(size_t)qrow * NN + kb0 + 16 + lm];
      float mx = fmaxf(s0, s1);
#pragma unroll
      for (int dms = 8; dms; dms >>= 1) mx = fmaxf(mx, __shfl_xor(mx, dms, 32));
      const float mnew = fmaxf(mrow[r], mx);
      const float alpha = __expf(mrow[r] - mnew);
      const float e0 = __expf(s0 - mnew), e1 = __expf(s1 - mnew);
      float sum = e0 + e1;
#pragma unroll
      for (int dms = 8; dms; dms >>= 1) sum += __shfl_xor(sum, dms, 32);
      lrow[r] = lrow[r] * alpha + sum;
      mrow[r] = mnew;
      p0[r] = e0; p1[r] = e1;
#pragma unroll
      for (int tt = 0; tt < 4; ++tt) o[tt][r] *= alpha;
    }

    // ---- P: C layout -> LDS -> A layout (bf16) ----
    bf16_t* pl = &plds[wave][0];
#pragma unroll
    for (int r = 0; r < 8; ++r) {
      const int row = r + half * 8;
      pl[row * 32 + lm]      = to_bf16(p0[r]);
      pl[row * 32 + 16 + lm] = to_bf16(p1[r]);
    }
    ABFrag pa;
    pa.h[0] = *(const bf16x8*)(pl + lm * 32 + half * 16);
    pa.h[1] = *(const bf16x8*)(pl + lm * 32 + half * 16 + 8);

    // ---- O += P V ----
#pragma unroll
    for (int tt = 0; tt < 4; ++tt)
      o[tt] = __builtin_amdgcn_wmma_f32_16x16x32_bf16(
          false, pa.v, false, vf[tt].v, (short)0, o[tt], false, false);

    if (more) {
#pragma unroll
      for (int j = 0; j < 2; ++j)
#pragma unroll
        for (int s = 0; s < 2; ++s) kf[j][s] = kf2[j][s];
    }
  }
#pragma unroll
  for (int tt = 0; tt < 4; ++tt) {
#pragma unroll
    for (int r = 0; r < 8; ++r) {
      const float val = o[tt][r] / lrow[r];
      const int qrow = q0 + r + half * 8;
      const int d = tt * 16 + lm;
      attn_a[((size_t)(b * NN + qrow)) * CC + h * DD + d] = to_bf16(val);
    }
  }
}

// ---------------------------------------------------------------------------
// proj GEMM + residual:  x1 = x + attn_a @ proj_w + proj_b   (fp32 out)
// ---------------------------------------------------------------------------
__global__ __launch_bounds__(256)
void gemm_proj_kernel(const bf16_t* __restrict__ A, const bf16_t* __restrict__ Wt,
                      const float* __restrict__ bias, const float* __restrict__ xin,
                      float* __restrict__ x1) {
  GEMM_LDS_DECL
  v8f acc[4] = {};
  GEMM_MAINLOOP(A, CC, Wt, CC, CC)
  const int row0 = blockIdx.y * 128 + wave * 16;
  const int col0 = blockIdx.x * 64;
#pragma unroll
  for (int tt = 0; tt < 4; ++tt) {
    const int col = col0 + tt * 16 + lm;
    const float bbv = bias[col];
#pragma unroll
    for (int r = 0; r < 8; ++r) {
      const int row = row0 + r + half * 8;
      const size_t idx = (size_t)row * CC + col;
      x1[idx] = xin[idx] + acc[tt][r] + bbv;
    }
  }
}

// ---------------------------------------------------------------------------
// fc1 GEMM + exact GELU -> bf16 [4096,3072]
// ---------------------------------------------------------------------------
__global__ __launch_bounds__(256)
void gemm_fc1_kernel(const bf16_t* __restrict__ A, const bf16_t* __restrict__ Wt,
                     const float* __restrict__ bias, bf16_t* __restrict__ hbuf) {
  GEMM_LDS_DECL
  v8f acc[4] = {};
  GEMM_MAINLOOP(A, CC, Wt, CC, CC)
  const int row0 = blockIdx.y * 128 + wave * 16;
  const int col0 = blockIdx.x * 64;
#pragma unroll
  for (int tt = 0; tt < 4; ++tt) {
    const int col = col0 + tt * 16 + lm;
    const float bbv = bias[col];
#pragma unroll
    for (int r = 0; r < 8; ++r) {
      const int row = row0 + r + half * 8;
      float v = acc[tt][r] + bbv;
      v = 0.5f * v * (1.0f + erff(v * 0.70710678118654752f));
      hbuf[(size_t)row * HID + col] = to_bf16(v);
    }
  }
}

// ---------------------------------------------------------------------------
// fc2 GEMM + residual -> fp32 d_out
// ---------------------------------------------------------------------------
__global__ __launch_bounds__(256)
void gemm_fc2_kernel(const bf16_t* __restrict__ A, const bf16_t* __restrict__ Wt,
                     const float* __restrict__ bias, const float* __restrict__ x1,
                     float* __restrict__ out) {
  GEMM_LDS_DECL
  v8f acc[4] = {};
  GEMM_MAINLOOP(A, HID, Wt, HID, HID)
  const int row0 = blockIdx.y * 128 + wave * 16;
  const int col0 = blockIdx.x * 64;
#pragma unroll
  for (int tt = 0; tt < 4; ++tt) {
    const int col = col0 + tt * 16 + lm;
    const float bbv = bias[col];
#pragma unroll
    for (int r = 0; r < 8; ++r) {
      const int row = row0 + r + half * 8;
      const size_t idx = (size_t)row * CC + col;
      out[idx] = x1[idx] + acc[tt][r] + bbv;
    }
  }
}

// ---------------------------------------------------------------------------
// Launch
// ---------------------------------------------------------------------------
extern "C" void kernel_launch(void* const* d_in, const int* in_sizes, int n_in,
                              void* d_out, int out_size, void* d_ws, size_t ws_size,
                              hipStream_t stream) {
  (void)in_sizes; (void)n_in; (void)out_size; (void)ws_size;
  const float* x      = (const float*)d_in[0];
  const float* pbias  = (const float*)d_in[1];
  const float* qkv_w  = (const float*)d_in[2];
  const float* qkv_b  = (const float*)d_in[3];
  const float* proj_w = (const float*)d_in[4];
  const float* proj_b = (const float*)d_in[5];
  const float* n1_w   = (const float*)d_in[6];
  const float* n1_b   = (const float*)d_in[7];
  const float* n2_w   = (const float*)d_in[8];
  const float* n2_b   = (const float*)d_in[9];
  const float* fc1_w  = (const float*)d_in[10];
  const float* fc1_b  = (const float*)d_in[11];
  const float* fc2_w  = (const float*)d_in[12];
  const float* fc2_b  = (const float*)d_in[13];
  float* out = (float*)d_out;

  char* wp = (char*)d_ws;
  auto alloc = [&](size_t bytes) {
    void* p = (void*)wp;
    wp += (bytes + 255) & ~(size_t)255;
    return p;
  };
  bf16_t* wt_qkv  = (bf16_t*)alloc((size_t)CC * 3 * CC * 2);
  bf16_t* wt_proj = (bf16_t*)alloc((size_t)CC * CC * 2);
  bf16_t* wt_fc1  = (bf16_t*)alloc((size_t)CC * HID * 2);
  bf16_t* wt_fc2  = (bf16_t*)alloc((size_t)HID * CC * 2);
  bf16_t* a1      = (bf16_t*)alloc((size_t)MTOT * CC * 2);
  bf16_t* qbuf    = (bf16_t*)alloc((size_t)BB * HH * NN * DD * 2);
  bf16_t* kbuf    = (bf16_t*)alloc((size_t)BB * HH * NN * DD * 2);
  bf16_t* vTbuf   = (bf16_t*)alloc((size_t)BB * HH * DD * NN * 2);
  bf16_t* attn_a  = (bf16_t*)alloc((size_t)MTOT * CC * 2);
  float*  x1      = (float*) alloc((size_t)MTOT * CC * 4);
  bf16_t* a2      = (bf16_t*)alloc((size_t)MTOT * CC * 2);
  bf16_t* hbuf    = (bf16_t*)alloc((size_t)MTOT * HID * 2);

  {
    int n;
    n = CC * 3 * CC; cvt_transpose_kernel<<<(n + 255) / 256, 256, 0, stream>>>(qkv_w,  wt_qkv,  CC,  3 * CC);
    n = CC * CC;     cvt_transpose_kernel<<<(n + 255) / 256, 256, 0, stream>>>(proj_w, wt_proj, CC,  CC);
    n = CC * HID;    cvt_transpose_kernel<<<(n + 255) / 256, 256, 0, stream>>>(fc1_w,  wt_fc1,  CC,  HID);
    n = HID * CC;    cvt_transpose_kernel<<<(n + 255) / 256, 256, 0, stream>>>(fc2_w,  wt_fc2,  HID, CC);
  }
  ln_bf16_kernel<<<MTOT, 256, 0, stream>>>(x, n1_w, n1_b, a1);
  gemm_qkv_kernel<<<dim3(3 * CC / 64, MTOT / 128), 256, 0, stream>>>(
      a1, wt_qkv, qkv_b, qbuf, kbuf, vTbuf);
  attn_kernel<<<dim3(NN / 64, BB * HH), 128, 0, stream>>>(
      qbuf, kbuf, vTbuf, pbias, attn_a);
  gemm_proj_kernel<<<dim3(CC / 64, MTOT / 128), 256, 0, stream>>>(
      attn_a, wt_proj, proj_b, x, x1);
  ln_bf16_kernel<<<MTOT, 256, 0, stream>>>(x1, n2_w, n2_b, a2);
  gemm_fc1_kernel<<<dim3(HID / 64, MTOT / 128), 256, 0, stream>>>(
      a2, wt_fc1, fc1_b, hbuf);
  gemm_fc2_kernel<<<dim3(CC / 64, MTOT / 128), 256, 0, stream>>>(
      hbuf, wt_fc2, fc2_b, x1, out);
}
